// generator_block_55860344652235
// MI455X (gfx1250) — matrix operand order repaired and checked by reading, not compiled, not run on hardware
//
#include <hip/hip_runtime.h>
#include <hip/hip_bf16.h>

typedef __attribute__((ext_vector_type(16))) _Float16 v16h;
typedef __attribute__((ext_vector_type(8)))  _Float16 v8h;
typedef __attribute__((ext_vector_type(8)))  float    v8f;

static constexpr int   CH   = 256;
static constexpr int   RES  = 128;
static constexpr float CONV_COEF  = 0.029462782549439483f; // sqrt(2)/sqrt(9*256)
static constexpr float DENSE_COEF = 0.0625f;               // sqrt(2)/sqrt(512)
static constexpr float IEPS = 1e-8f;

// ---------------------------------------------------------------- utilities

__global__ void zero_stats_kernel(float* stats) {
  int i = blockIdx.x * 256 + threadIdx.x;
  if (i < 4 * 2048) stats[i] = 0.f;
}

// conv_w layout [ky][kx][ci][co] f32  ->  wT [tap][co][ci] f16, pre-scaled.
__global__ void prep_weights_kernel(const float* __restrict__ w0,
                                    const float* __restrict__ w1,
                                    _Float16* __restrict__ wT0,
                                    _Float16* __restrict__ wT1) {
  int i = blockIdx.x * 256 + threadIdx.x;       // over 9*256*256, co fastest
  if (i >= 9 * CH * CH) return;
  int co  = i & 255;
  int ci  = (i >> 8) & 255;
  int tap = i >> 16;
  wT0[(tap * CH + co) * CH + ci] = (_Float16)(w0[(tap * CH + ci) * CH + co] * CONV_COEF);
  wT1[(tap * CH + co) * CH + ci] = (_Float16)(w1[(tap * CH + ci) * CH + co] * CONV_COEF);
}

// style = w[b,layer] @ (dense_w * DENSE_COEF) + dense_b ; split into ys/yb.
__global__ void style_kernel(const float* __restrict__ w,
                             const float* __restrict__ d0w, const float* __restrict__ d0b,
                             const float* __restrict__ d1w, const float* __restrict__ d1b,
                             float* ys0, float* yb0, float* ys1, float* yb1) {
  int blk = blockIdx.x;                 // 16 blocks: b*2 + layer
  int b = blk >> 1, layer = blk & 1;
  const float* dw = layer ? d1w : d0w;
  const float* db = layer ? d1b : d0b;
  const float* wv = w + (b * 2 + layer) * 512;
  int j = threadIdx.x;                  // 0..255
  float a0 = 0.f, a1 = 0.f;
  for (int k = 0; k < 512; ++k) {
    float wk = wv[k];
    a0 += wk * dw[k * 512 + j];
    a1 += wk * dw[k * 512 + j + 256];
  }
  float* ys = layer ? ys1 : ys0;
  float* yb = layer ? yb1 : yb0;
  ys[b * CH + j] = a0 * DENSE_COEF + db[j];
  yb[b * CH + j] = a1 * DENSE_COEF + db[j + 256];
}

// per (b,c): instance-norm + style -> affine scale/bias
__global__ void adain_params_kernel(const float* sum, const float* sumsq,
                                    const float* ys, const float* yb,
                                    float* sc, float* bi) {
  int i = blockIdx.x * 256 + threadIdx.x;
  if (i >= 2048) return;
  const float inv = 1.f / (RES * RES);
  float mu  = sum[i] * inv;
  float var = sumsq[i] * inv - mu * mu;
  float rs  = rsqrtf(var + IEPS);
  float s   = (1.f + ys[i]) * rs;
  sc[i] = s;
  bi[i] = yb[i] - s * mu;
}

// ------------------------------------------------------------- staging helpers

// conv0 input slab: 34 cols x 256 ch of upsampled(x) row gy, f32->f16, zero-pad.
__device__ __forceinline__ void stage_conv0(_Float16* __restrict__ dst,
                                            const float* __restrict__ x,
                                            int b, int gy, int ox0, int tid) {
  for (int idx = tid; idx < 34 * CH; idx += 256) {
    int col = idx >> 8, ch = idx & 255;
    int gx = ox0 + col - 1;
    float v = 0.f;
    if ((unsigned)gy < 128u && (unsigned)gx < 128u)
      v = x[((b * 64 + (gy >> 1)) * 64 + (gx >> 1)) * CH + ch];
    dst[idx] = (_Float16)v;
  }
}

// conv1 input slab: AdaIN-0 applied on load; SAME padding stays zero.
__device__ __forceinline__ void stage_conv1(_Float16* __restrict__ dst,
                                            const _Float16* __restrict__ h0,
                                            const float* __restrict__ sc0,
                                            const float* __restrict__ bi0,
                                            int b, int gy, int ox0, int tid) {
  for (int idx = tid; idx < 34 * CH; idx += 256) {
    int col = idx >> 8, ch = idx & 255;
    int gx = ox0 + col - 1;
    float v = 0.f;
    if ((unsigned)gy < 128u && (unsigned)gx < 128u) {
      float hv = (float)h0[(((long)b * RES + gy) * RES + gx) * CH + ch];
      v = hv * sc0[b * CH + ch] + bi0[b * CH + ch];
    }
    dst[idx] = (_Float16)v;
  }
}

// --------------------------------------------------------------- WMMA tap core
// One (ky,kx) tap: 8 K-chunks of 32, 4 N-tiles per wave, B software-pipelined
// one chunk ahead in registers so each WMMA depends only on loads issued a full
// iteration earlier (keeps >=8 loads in flight instead of wait-loadcnt-0).

__device__ __forceinline__ void tap_mma(const _Float16* __restrict__ arow,
                                        const _Float16* __restrict__ bbase,
                                        v8f acc[4]) {
  __builtin_prefetch(bbase + CH * CH, 0, 1);      // warm next tap's weight row
  v16h Bc[4], Bn[4];
#pragma unroll
  for (int t = 0; t < 4; ++t)
    Bc[t] = *(const v16h*)(bbase + t * 16 * CH);
#pragma unroll
  for (int kc = 0; kc < 8; ++kc) {
    if (kc < 7) {
#pragma unroll
      for (int t = 0; t < 4; ++t)
        Bn[t] = *(const v16h*)(bbase + t * 16 * CH + (kc + 1) * 32);
    }
    v8h a0 = *(const v8h*)(arow + kc * 32);
    v8h a1 = *(const v8h*)(arow + kc * 32 + 8);
    v16h A = __builtin_shufflevector(a0, a1, 0,1,2,3,4,5,6,7,8,9,10,11,12,13,14,15);
#pragma unroll
    for (int t = 0; t < 4; ++t)
      acc[t] = __builtin_amdgcn_wmma_f32_16x16x32_f16(false, A, false, Bc[t],
                                                      (short)0, acc[t], false, false);
#pragma unroll
    for (int t = 0; t < 4; ++t)
      Bc[t] = Bn[t];
  }
}

// ------------------------------------------------------- conv0 (upsample fused)
// WG = 32 pixels x 256 out-ch. 8 waves: p in {0,1} pixel half, c4 in {0..3} gives
// 64 out-ch per wave = 4 N-tiles -> one A fragment feeds 4 WMMAs.

__global__ __launch_bounds__(256)
void conv0_wmma_kernel(const float* __restrict__ x, const _Float16* __restrict__ wT,
                       _Float16* __restrict__ h0raw) {
  __shared__ _Float16 tile[2][34 * CH];         // double-buffered row slabs
  int wg = blockIdx.x;                          // 4096 = 4xt * 128oy * 8b
  int xt = wg & 3, oy = (wg >> 2) & 127, b = wg >> 9;
  int tid = threadIdx.x, lane = tid & 31, wave = tid >> 5;
  int p = wave & 1, c4 = wave >> 1;
  int ox0 = xt * 32;
  int m = lane & 15, hi = lane >> 4, n = lane & 15;
  v8f acc[4] = {};
  stage_conv0(tile[0], x, b, oy - 1, ox0, tid);
  __syncthreads();
  for (int ky = 0; ky < 3; ++ky) {
    if (ky < 2) stage_conv0(tile[(ky + 1) & 1], x, b, oy + ky, ox0, tid);
    for (int kx = 0; kx < 3; ++kx) {
      const _Float16* arow  = &tile[ky & 1][(p * 16 + m + kx) * CH + hi * 16];
      const _Float16* bbase = &wT[((ky * 3 + kx) * CH + c4 * 64 + n) * CH + hi * 16];
      tap_mma(arow, bbase, acc);
    }
    __syncthreads();
  }
  long rowbase = ((long)(b * RES + oy) * RES + ox0 + p * 16) * CH;
#pragma unroll
  for (int t = 0; t < 4; ++t) {
    int co = c4 * 64 + t * 16 + n;
#pragma unroll
    for (int r = 0; r < 8; ++r)
      h0raw[rowbase + (long)(r + 8 * hi) * CH + co] = (_Float16)acc[t][r];
  }
}

// ------------------------------- blur + noise + bias + lrelu + IN stats (fused)

__global__ __launch_bounds__(256)
void blur_noise_stats_kernel(const _Float16* __restrict__ h0raw,
                             const float* __restrict__ noise,
                             const float* __restrict__ ns0, const float* __restrict__ b0,
                             _Float16* __restrict__ h0, float* sum0, float* sumsq0) {
  int blk = blockIdx.x;                 // 1024 = 8b * 128 rows
  int b = blk >> 7, y = blk & 127;
  int ch = threadIdx.x;
  float nsc = ns0[ch], bs = b0[ch];
  float s = 0.f, s2 = 0.f;
  for (int xx = 0; xx < RES; ++xx) {
    float a = 0.f;
#pragma unroll
    for (int dy = -1; dy <= 1; ++dy) {
      int yy = y + dy; if ((unsigned)yy >= 128u) continue;
#pragma unroll
      for (int dx = -1; dx <= 1; ++dx) {
        int xq = xx + dx; if ((unsigned)xq >= 128u) continue;
        float wgt = ((dy == 0) ? 2.f : 1.f) * ((dx == 0) ? 2.f : 1.f) * (1.f / 16.f);
        a += wgt * (float)h0raw[(((long)b * RES + yy) * RES + xq) * CH + ch];
      }
    }
    float nz = noise[(((long)b * RES + y) * RES + xx) * 2 + 0];
    float v = a + nz * nsc + bs;
    v = v > 0.f ? v : 0.2f * v;
    h0[(((long)b * RES + y) * RES + xx) * CH + ch] = (_Float16)v;
    s += v; s2 += v * v;
  }
  atomicAdd(&sum0[b * CH + ch], s);
  atomicAdd(&sumsq0[b * CH + ch], s2);
}

// --------------------------- conv1 (AdaIN-0 fused on load, epilogue fused stats)

__global__ __launch_bounds__(256)
void conv1_wmma_kernel(const _Float16* __restrict__ h0, const _Float16* __restrict__ wT,
                       const float* __restrict__ sc0, const float* __restrict__ bi0,
                       const float* __restrict__ noise,
                       const float* __restrict__ ns1, const float* __restrict__ b1,
                       _Float16* __restrict__ h1, float* sum1, float* sumsq1) {
  __shared__ _Float16 tile[2][34 * CH];
  __shared__ float rsum[256], rsum2[256];
  int wg = blockIdx.x;                          // 4096
  int xt = wg & 3, oy = (wg >> 2) & 127, b = wg >> 9;
  int tid = threadIdx.x, lane = tid & 31, wave = tid >> 5;
  int p = wave & 1, c4 = wave >> 1;
  int ox0 = xt * 32;
  int m = lane & 15, hi = lane >> 4, n = lane & 15;
  rsum[tid] = 0.f; rsum2[tid] = 0.f;
  v8f acc[4] = {};
  stage_conv1(tile[0], h0, sc0, bi0, b, oy - 1, ox0, tid);
  __syncthreads();
  for (int ky = 0; ky < 3; ++ky) {
    if (ky < 2) stage_conv1(tile[(ky + 1) & 1], h0, sc0, bi0, b, oy + ky, ox0, tid);
    for (int kx = 0; kx < 3; ++kx) {
      const _Float16* arow  = &tile[ky & 1][(p * 16 + m + kx) * CH + hi * 16];
      const _Float16* bbase = &wT[((ky * 3 + kx) * CH + c4 * 64 + n) * CH + hi * 16];
      tap_mma(arow, bbase, acc);
    }
    __syncthreads();
  }
  long pixbase = (long)(b * RES + oy) * RES + ox0 + p * 16;
  float nz[8];
#pragma unroll
  for (int r = 0; r < 8; ++r)
    nz[r] = noise[(pixbase + r + 8 * hi) * 2 + 1];
#pragma unroll
  for (int t = 0; t < 4; ++t) {
    int co = c4 * 64 + t * 16 + n;
    float nsc = ns1[co], bs = b1[co];
    float s = 0.f, s2 = 0.f;
#pragma unroll
    for (int r = 0; r < 8; ++r) {
      float v = acc[t][r] + nz[r] * nsc + bs;
      v = v > 0.f ? v : 0.2f * v;
      h1[(pixbase + r + 8 * hi) * CH + co] = (_Float16)v;
      s += v; s2 += v * v;
    }
    atomicAdd(&rsum[co], s);
    atomicAdd(&rsum2[co], s2);
  }
  __syncthreads();
  atomicAdd(&sum1[b * CH + tid], rsum[tid]);
  atomicAdd(&sumsq1[b * CH + tid], rsum2[tid]);
}

// ---------------------------------------------------------------- final AdaIN

__global__ void final_out_kernel(const _Float16* __restrict__ h1,
                                 const float* __restrict__ sc1,
                                 const float* __restrict__ bi1,
                                 float* __restrict__ out) {
  long i = (long)blockIdx.x * 256 + threadIdx.x;   // 8*128*128*256
  int ch = (int)(i & 255);
  int b  = (int)(i >> 22);                          // 128*128*256 = 2^22
  out[i] = (float)h1[i] * sc1[b * CH + ch] + bi1[b * CH + ch];
}

// ------------------------------------------------------------------ launcher

extern "C" void kernel_launch(void* const* d_in, const int* in_sizes, int n_in,
                              void* d_out, int out_size, void* d_ws, size_t ws_size,
                              hipStream_t stream) {
  const float* x    = (const float*)d_in[0];
  const float* w    = (const float*)d_in[1];
  const float* nois = (const float*)d_in[2];
  const float* c0w  = (const float*)d_in[3];
  const float* c1w  = (const float*)d_in[4];
  const float* ns0  = (const float*)d_in[5];
  const float* ns1  = (const float*)d_in[6];
  const float* b0   = (const float*)d_in[7];
  const float* b1   = (const float*)d_in[8];
  const float* d0w  = (const float*)d_in[9];
  const float* d0b  = (const float*)d_in[10];
  const float* d1w  = (const float*)d_in[11];
  const float* d1b  = (const float*)d_in[12];

  char* ws = (char*)d_ws;
  float* sum0   = (float*)(ws + 0);
  float* sumsq0 = (float*)(ws + 8192);
  float* sum1   = (float*)(ws + 16384);
  float* sumsq1 = (float*)(ws + 24576);
  float* ys0    = (float*)(ws + 32768);
  float* yb0    = (float*)(ws + 40960);
  float* ys1    = (float*)(ws + 49152);
  float* yb1    = (float*)(ws + 57344);
  float* sc0    = (float*)(ws + 65536);
  float* bi0    = (float*)(ws + 73728);
  float* sc1    = (float*)(ws + 81920);
  float* bi1    = (float*)(ws + 90112);
  _Float16* wT0   = (_Float16*)(ws + (1u << 17));
  _Float16* wT1   = (_Float16*)(ws + (1u << 17) + 9 * 256 * 256 * 2);
  _Float16* h0raw = (_Float16*)(ws + (4u << 20));                    // 64 MB
  _Float16* h0    = (_Float16*)(ws + (4u << 20) + (64u << 20));      // 64 MB
  _Float16* h1    = h0raw;   // h0raw fully consumed by blur before conv1 writes h1

  zero_stats_kernel<<<32, 256, 0, stream>>>(sum0);
  prep_weights_kernel<<<(9 * 256 * 256 + 255) / 256, 256, 0, stream>>>(c0w, c1w, wT0, wT1);
  style_kernel<<<16, 256, 0, stream>>>(w, d0w, d0b, d1w, d1b, ys0, yb0, ys1, yb1);
  conv0_wmma_kernel<<<4096, 256, 0, stream>>>(x, wT0, h0raw);
  blur_noise_stats_kernel<<<1024, 256, 0, stream>>>(h0raw, nois, ns0, b0, h0, sum0, sumsq0);
  adain_params_kernel<<<8, 256, 0, stream>>>(sum0, sumsq0, ys0, yb0, sc0, bi0);
  conv1_wmma_kernel<<<4096, 256, 0, stream>>>(h0, wT1, sc0, bi0, nois, ns1, b1,
                                              h1, sum1, sumsq1);
  adain_params_kernel<<<8, 256, 0, stream>>>(sum1, sumsq1, ys1, yb1, sc1, bi1);
  final_out_kernel<<<131072, 256, 0, stream>>>(h1, sc1, bi1, (float*)d_out);
}
